// MultiHeadAttention_51238959841967
// MI455X (gfx1250) — matrix operand order repaired and checked by reading, not compiled, not run on hardware
//
#include <hip/hip_runtime.h>
#include <hip/hip_bf16.h>

// ---------------------------------------------------------------------------
// MultiHeadAttention forward for gfx1250 (MI455X), wave32 + WMMA f16 + TDM.
// MODEL=1024, H=16, DK=64, B=2, S=2048. Outputs: [out (B,S,1024) | w (32,S,S)].
// ---------------------------------------------------------------------------

typedef __attribute__((ext_vector_type(16))) _Float16     v16h;
typedef __attribute__((ext_vector_type(8)))  float        v8f;
typedef __attribute__((ext_vector_type(4)))  unsigned int v4u;
typedef __attribute__((ext_vector_type(8)))  int          v8i;
typedef __attribute__((ext_vector_type(4)))  int          v4i;

union V16H { v16h v; _Float16 e[16]; };
union V8F  { v8f  v; float    e[8];  };

__device__ inline v8f zero_v8f() {
  v8f z = {0.f, 0.f, 0.f, 0.f, 0.f, 0.f, 0.f, 0.f};
  return z;
}

__device__ inline v8f wmma_f16(v16h a, v16h b, v8f c) {
  // (neg_a, A, neg_b, B, c_mod, C, reuse_a, reuse_b) -> v_wmma_f32_16x16x32_f16
  return __builtin_amdgcn_wmma_f32_16x16x32_f16(false, a, false, b, (short)0, c,
                                                false, false);
}

// Truncating a generic pointer to shared memory yields the LDS byte offset
// (ISA 10.2: LDS aperture, low 32 address bits).
__device__ inline unsigned lds_off(const void* p) {
  return (unsigned)(unsigned long long)p;
}

// ---------------------------------------------------------------------------
// Tensor Data Mover: 2D tile load Global -> LDS (f16 elements, data_size=2B).
// Builds D# per CDNA5 ISA 8.3/8.4: group0 {count=1, lds_addr, global_addr,
// type=2}; group1 {data_size=1(2B), tensor_dim0/1, tile_dim0/1, dim0 stride}.
// Tile rows are packed contiguously in LDS (row-major [tile_d1][tile_d0]).
// ---------------------------------------------------------------------------
__device__ inline void tdm_load_2d_f16(unsigned lds_addr, const void* gptr,
                                       unsigned tile_d0,   // elems per row
                                       unsigned tile_d1,   // rows
                                       unsigned long long stride0_elems) {
  unsigned long long ga = (unsigned long long)gptr;
  v4u g0;
  g0[0] = 1u;                                        // count=1, user D#
  g0[1] = lds_addr;                                  // lds_addr [63:32]
  g0[2] = (unsigned)(ga & 0xffffffffu);              // global_addr lo
  g0[3] = (unsigned)((ga >> 32) & 0x01ffffffu) | (2u << 30);  // addr hi | type=2

  const unsigned td0 = tile_d0;        // tensor_dim0 == tile width (no OOB)
  const unsigned td1 = tile_d1;        // tensor_dim1 == tile rows  (no OOB)
  v8i g1;
  g1[0] = (int)(1u << 16);             // workgroup_mask=0, data_size=1 (2B)
  g1[1] = (int)((td0 & 0xffffu) << 16);                   // tensor_dim0 lo16
  g1[2] = (int)((td0 >> 16) | ((td1 & 0xffffu) << 16));   // d0 hi | d1 lo
  g1[3] = (int)((td1 >> 16) | ((tile_d0 & 0xffffu) << 16)); // d1 hi | tile_dim0
  g1[4] = (int)(tile_d1 & 0xffffu);                       // tile_dim1 (dim2=0)
  g1[5] = (int)(stride0_elems & 0xffffffffu);             // dim0_stride lo32
  g1[6] = (int)((stride0_elems >> 32) & 0xffffu);         // dim0_stride hi16
  g1[7] = 0;                                              // dim1_stride (2D)
  v4i g2 = {0, 0, 0, 0};
  v4i g3 = {0, 0, 0, 0};
#if defined(__clang_major__) && __clang_major__ >= 23
  v8i g4 = {0, 0, 0, 0, 0, 0, 0, 0};
  __builtin_amdgcn_tensor_load_to_lds(g0, g1, g2, g3, g4, 0);
#else
  __builtin_amdgcn_tensor_load_to_lds(g0, g1, g2, g3, 0);
#endif
}

// A fragment: 16x32 f16, row-major source [m][k] with row stride `stride`.
__device__ inline v16h load_a_frag(const _Float16* base, int stride, int m_off) {
  const int lane = threadIdx.x & 31;
  const int half = lane >> 4;
  const _Float16* rp = base + (m_off + (lane & 15)) * stride;
  V16H f;
#pragma unroll
  for (int v = 0; v < 4; ++v) {
    int k0 = half * 8 + 2 * v;
    f.e[2 * v]     = rp[k0];
    f.e[2 * v + 1] = rp[k0 + 1];
    int k1 = 16 + half * 8 + 2 * v;
    f.e[8 + 2 * v]     = rp[k1];
    f.e[8 + 2 * v + 1] = rp[k1 + 1];
  }
  return f.v;
}

// B fragment: 32x16 f16, source [k][n] row-major with row stride `stride`.
__device__ inline v16h load_b_frag(const _Float16* base, int stride, int n_off) {
  const int lane = threadIdx.x & 31;
  const int half = lane >> 4;
  const _Float16* rp = base + n_off + (lane & 15);
  V16H f;
#pragma unroll
  for (int i = 0; i < 8; ++i) {
    f.e[i]     = rp[(half * 8 + i) * stride];
    f.e[8 + i] = rp[(16 + half * 8 + i) * stride];
  }
  return f.v;
}

// ---------------------------------------------------------------------------
// Kernel 0: fp32 -> f16 weight conversion
// ---------------------------------------------------------------------------
__global__ __launch_bounds__(256) void mha_cvt_f16(const float* __restrict__ src,
                                                   _Float16* __restrict__ dst,
                                                   int n) {
  int i = blockIdx.x * 256 + threadIdx.x;
  if (i < n) dst[i] = (_Float16)src[i];
}

// ---------------------------------------------------------------------------
// Kernel 1: projection  P = X @ Wq^T + bq  -> head-split f16 [B*H][S][64]
// ---------------------------------------------------------------------------
__global__ __launch_bounds__(128) void mha_proj(const float* __restrict__ X,
                                                const _Float16* __restrict__ W,
                                                const float* __restrict__ bias,
                                                _Float16* __restrict__ dst) {
  __shared__ _Float16 As[64][32];  // [m][k]
  __shared__ _Float16 Bs[32][64];  // [k][n]
  const int tid = threadIdx.x;
  const int wave = tid >> 5;
  const int lane = tid & 31;
  const int m0 = blockIdx.x * 64;
  const int n0 = blockIdx.y * 64;

  V8F acc[4];
#pragma unroll
  for (int i = 0; i < 4; ++i) acc[i].v = zero_v8f();

  for (int k0 = 0; k0 < 1024; k0 += 32) {
    __syncthreads();
    {  // stage A (fp32 -> f16): thread t -> row=t&63, khalf=t>>6, 16 elems
      int row = tid & 63, kh = tid >> 6;
      const float* src = X + (size_t)(m0 + row) * 1024 + k0 + kh * 16;
      __builtin_prefetch(src + 32, 0, 0);  // next K tile -> global_prefetch_b8
      _Float16* d = &As[row][kh * 16];
#pragma unroll
      for (int i = 0; i < 16; ++i) d[i] = (_Float16)src[i];
    }
    {  // stage B transposed: Bs[k][n] = W[n0+n][k0+k]
      int n = tid & 63, kh = tid >> 6;
      const _Float16* src = W + (size_t)(n0 + n) * 1024 + k0 + kh * 16;
#pragma unroll
      for (int i = 0; i < 16; ++i) Bs[kh * 16 + i][n] = src[i];
    }
    __syncthreads();
    v16h a = load_a_frag(&As[0][0], 32, wave * 16);
#pragma unroll
    for (int nb = 0; nb < 4; ++nb) {
      v16h b = load_b_frag(&Bs[0][0], 64, nb * 16);
      acc[nb].v = wmma_f16(a, b, acc[nb].v);
    }
  }

  const int half = lane >> 4;
  const int nidx = lane & 15;
#pragma unroll
  for (int nb = 0; nb < 4; ++nb) {
#pragma unroll
    for (int r = 0; r < 8; ++r) {
      int row = m0 + wave * 16 + r + 8 * half;  // 0..4095 (b*2048+s)
      int col = n0 + nb * 16 + nidx;            // 0..1023 (h*64+dk)
      float v = acc[nb].e[r] + bias[col];
      int b = row >> 11, s = row & 2047;
      int h = col >> 6,  dk = col & 63;
      dst[(((size_t)(b * 16 + h) * 2048) + s) * 64 + dk] = (_Float16)v;
    }
  }
}

// ---------------------------------------------------------------------------
// Kernel 2: fused attention. One workgroup = (bh, 64 rows), 128 threads.
// ---------------------------------------------------------------------------
__device__ inline void mha_stage_k(const _Float16* kp, int bh, int key0,
                                   _Float16 (*Bt)[32]) {
  int t = threadIdx.x;
  int key = t & 31, kh = t >> 5;  // kh 0..3
  const _Float16* src = kp + ((size_t)(bh * 2048 + key0 + key)) * 64 + kh * 16;
#pragma unroll
  for (int i = 0; i < 16; ++i) Bt[kh * 16 + i][key] = src[i];  // [dk][key]
}

__device__ inline v8f mha_score(v16h qa0, v16h qa1, const _Float16 (*Bt)[32],
                                int sub) {
  v8f sc = zero_v8f();
  sc = wmma_f16(qa0, load_b_frag(&Bt[0][0],  32, sub * 16), sc);  // dk 0..31
  sc = wmma_f16(qa1, load_b_frag(&Bt[32][0], 32, sub * 16), sc);  // dk 32..63
  return sc;
}

__global__ __launch_bounds__(128) void mha_attn(const _Float16* __restrict__ qp,
                                                const _Float16* __restrict__ kp,
                                                const _Float16* __restrict__ vp,
                                                const unsigned char* __restrict__ mask,
                                                float* __restrict__ w_out,
                                                _Float16* __restrict__ ctx) {
  __shared__ _Float16 Bt[64][32];       // kp tile  [dk][key]   (32 keys)
  __shared__ _Float16 Vt[32][64];       // vp tile  [key][dk]   (TDM-loaded)
  __shared__ _Float16 Wt[4][16][32];    // per-wave w staging [m][key]

  const int tid = threadIdx.x, wave = tid >> 5, lane = tid & 31;
  const int half = lane >> 4, nidx = lane & 15;
  const int bh = blockIdx.y;
  const int rows0 = blockIdx.x * 64;
  const int mrow = rows0 + wave * 16;          // seq row base of this wave
  const int mb = bh & 1;                       // tile(mask,(H,1,1)) quirk: (b*H+h)%B
  const float SCALE = 0.03125f;                // 1/sqrt(MODEL_SIZE) quirk

  const _Float16* qbase = qp + ((size_t)(bh * 2048 + mrow)) * 64;
  v16h qa0 = load_a_frag(qbase, 64, 0);        // dk 0..31
  v16h qa1 = load_a_frag(qbase + 32, 64, 0);   // dk 32..63

  float rowmax[8], rowsum[8];
#pragma unroll
  for (int r = 0; r < 8; ++r) { rowmax[r] = -3.0e38f; rowsum[r] = 0.f; }

  // ---------------- pass 1: stats ----------------
  for (int kt = 0; kt < 64; ++kt) {
    __syncthreads();
    mha_stage_k(kp, bh, kt * 32, Bt);
    __syncthreads();
#pragma unroll
    for (int sub = 0; sub < 2; ++sub) {
      V8F sc; sc.v = mha_score(qa0, qa1, Bt, sub);
      int col = kt * 32 + sub * 16 + nidx;
#pragma unroll
      for (int r = 0; r < 8; ++r) {
        int srow = mrow + r + 8 * half;
        float s = sc.e[r] * SCALE;
        if (mask[((size_t)(mb * 2048 + srow)) * 2048 + col]) s = -1.0e18f;
        if (s > rowmax[r]) {
          rowsum[r] *= __expf(rowmax[r] - s);
          rowmax[r] = s;
        }
        rowsum[r] += __expf(s - rowmax[r]);
      }
    }
  }
  // reduce stats across the 16 lanes of each half-wave
#pragma unroll
  for (int off = 1; off < 16; off <<= 1) {
#pragma unroll
    for (int r = 0; r < 8; ++r) {
      float om = __shfl_xor(rowmax[r], off);
      float os = __shfl_xor(rowsum[r], off);
      float nm = fmaxf(rowmax[r], om);
      rowsum[r] = rowsum[r] * __expf(rowmax[r] - nm) + os * __expf(om - nm);
      rowmax[r] = nm;
    }
  }
  float rinv[8];
#pragma unroll
  for (int r = 0; r < 8; ++r) rinv[r] = 1.0f / rowsum[r];

  // ---------------- pass 2: w + context ----------------
  V8F oacc[4];
#pragma unroll
  for (int i = 0; i < 4; ++i) oacc[i].v = zero_v8f();

  const unsigned vt_lds = lds_off(&Vt[0][0]);
  for (int kt = 0; kt < 64; ++kt) {
    __syncthreads();  // all waves done reading previous Bt/Vt
    if (wave == 0) {  // TDM: vp tile (32 keys x 64 dk f16, contiguous rows)
      tdm_load_2d_f16(vt_lds, vp + ((size_t)(bh * 2048 + kt * 32)) * 64,
                      /*tile_d0=*/64, /*tile_d1=*/32, /*stride0=*/64);
    }
    mha_stage_k(kp, bh, kt * 32, Bt);
    if (wave == 0) __builtin_amdgcn_s_wait_tensorcnt((short)0);
    __syncthreads();
#pragma unroll
    for (int sub = 0; sub < 2; ++sub) {
      V8F sc; sc.v = mha_score(qa0, qa1, Bt, sub);
      int col = kt * 32 + sub * 16 + nidx;
#pragma unroll
      for (int r = 0; r < 8; ++r) {
        int srow = mrow + r + 8 * half;
        float s = sc.e[r] * SCALE;
        if (mask[((size_t)(mb * 2048 + srow)) * 2048 + col]) s = -1.0e18f;
        float w = __expf(s - rowmax[r]) * rinv[r];  // final normalized weight
        // w is write-once, never re-read: non-temporal keeps L2 for kp/vp/mask
        __builtin_nontemporal_store(
            w, &w_out[((size_t)(bh * 2048 + srow)) * 2048 + col]);
        Wt[wave][r + 8 * half][sub * 16 + nidx] = (_Float16)w;
      }
    }
    // context += w(16x32) @ vp(32x64); Wt slice is wave-private (LDS in-order)
    v16h wa = load_a_frag(&Wt[wave][0][0], 32, 0);
#pragma unroll
    for (int nb = 0; nb < 4; ++nb)
      oacc[nb].v = wmma_f16(wa, load_b_frag(&Vt[0][0], 64, nb * 16), oacc[nb].v);
  }

  const int b = bh >> 4, h = bh & 15;
#pragma unroll
  for (int nb = 0; nb < 4; ++nb) {
#pragma unroll
    for (int r = 0; r < 8; ++r) {
      int srow = mrow + r + 8 * half;
      int dk = nb * 16 + nidx;
      ctx[((size_t)(b * 2048 + srow)) * 1024 + h * 64 + dk] =
          (_Float16)oacc[nb].e[r];
    }
  }
}

// ---------------------------------------------------------------------------
// Kernel 3: output projection  Y = ctx @ Wout^T + bout + residual(q)  (fp32)
// A tile (f16, 64 rows x 32 cols, stride 1024) staged by TDM.
// ---------------------------------------------------------------------------
__global__ __launch_bounds__(128) void mha_outproj(const _Float16* __restrict__ Xh,
                                                   const _Float16* __restrict__ W,
                                                   const float* __restrict__ bias,
                                                   const float* __restrict__ resid,
                                                   float* __restrict__ Y) {
  __shared__ _Float16 As[64][32];
  __shared__ _Float16 Bs[32][64];
  const int tid = threadIdx.x;
  const int wave = tid >> 5;
  const int lane = tid & 31;
  const int m0 = blockIdx.x * 64;
  const int n0 = blockIdx.y * 64;

  V8F acc[4];
#pragma unroll
  for (int i = 0; i < 4; ++i) acc[i].v = zero_v8f();

  const unsigned as_lds = lds_off(&As[0][0]);
  for (int k0 = 0; k0 < 1024; k0 += 32) {
    __syncthreads();
    if (wave == 0) {  // TDM: A tile 64 rows x 32 f16, row stride 1024 elems
      tdm_load_2d_f16(as_lds, Xh + (size_t)m0 * 1024 + k0,
                      /*tile_d0=*/32, /*tile_d1=*/64, /*stride0=*/1024);
    }
    {
      int n = tid & 63, kh = tid >> 6;
      const _Float16* src = W + (size_t)(n0 + n) * 1024 + k0 + kh * 16;
      __builtin_prefetch(src + 32, 0, 0);
#pragma unroll
      for (int i = 0; i < 16; ++i) Bs[kh * 16 + i][n] = src[i];
    }
    if (wave == 0) __builtin_amdgcn_s_wait_tensorcnt((short)0);
    __syncthreads();
    v16h a = load_a_frag(&As[0][0], 32, wave * 16);
#pragma unroll
    for (int nb = 0; nb < 4; ++nb) {
      v16h b = load_b_frag(&Bs[0][0], 64, nb * 16);
      acc[nb].v = wmma_f16(a, b, acc[nb].v);
    }
  }

  const int half = lane >> 4;
  const int nidx = lane & 15;
#pragma unroll
  for (int nb = 0; nb < 4; ++nb) {
#pragma unroll
    for (int r = 0; r < 8; ++r) {
      int row = m0 + wave * 16 + r + 8 * half;
      int col = n0 + nb * 16 + nidx;
      size_t idx = (size_t)row * 1024 + col;
      Y[idx] = acc[nb].e[r] + bias[col] + resid[idx];
    }
  }
}

// ---------------------------------------------------------------------------
// Kernel 4: row LayerNorm (1024 cols), eps=1e-6
// ---------------------------------------------------------------------------
__global__ __launch_bounds__(256) void mha_layernorm(const float* __restrict__ Y,
                                                     const float* __restrict__ gamma,
                                                     const float* __restrict__ beta,
                                                     float* __restrict__ out) {
  __shared__ float ps[8], ps2[8];
  const int row = blockIdx.x;
  const float* x = Y + (size_t)row * 1024;
  float s = 0.f, s2 = 0.f;
  for (int i = threadIdx.x; i < 1024; i += 256) {
    float v = x[i];
    s += v; s2 += v * v;
  }
#pragma unroll
  for (int off = 16; off > 0; off >>= 1) {
    s  += __shfl_down(s, off);
    s2 += __shfl_down(s2, off);
  }
  int wv = threadIdx.x >> 5, ln = threadIdx.x & 31;
  if (ln == 0) { ps[wv] = s; ps2[wv] = s2; }
  __syncthreads();
  float tot = 0.f, tot2 = 0.f;
#pragma unroll
  for (int i = 0; i < 8; ++i) { tot += ps[i]; tot2 += ps2[i]; }
  float mean = tot * (1.0f / 1024.0f);
  float var  = tot2 * (1.0f / 1024.0f) - mean * mean;
  float inv  = rsqrtf(var + 1e-6f);
  for (int i = threadIdx.x; i < 1024; i += 256) {
    out[(size_t)row * 1024 + i] = (x[i] - mean) * inv * gamma[i] + beta[i];
  }
}

// ---------------------------------------------------------------------------
// Host launcher
// ---------------------------------------------------------------------------
extern "C" void kernel_launch(void* const* d_in, const int* in_sizes, int n_in,
                              void* d_out, int out_size, void* d_ws, size_t ws_size,
                              hipStream_t stream) {
  (void)in_sizes; (void)n_in; (void)out_size; (void)ws_size;

  const float* q    = (const float*)d_in[0];
  const float* k    = (const float*)d_in[1];
  const float* v    = (const float*)d_in[2];
  const unsigned char* mask = (const unsigned char*)d_in[3];
  const float* Wq   = (const float*)d_in[4];
  const float* bq   = (const float*)d_in[5];
  const float* Wout = (const float*)d_in[6];
  const float* bout = (const float*)d_in[7];
  const float* gamma = (const float*)d_in[8];
  const float* beta  = (const float*)d_in[9];

  // workspace layout
  _Float16* Wq16 = (_Float16*)d_ws;                 // 1M f16
  _Float16* Wo16 = Wq16 + (size_t)1024 * 1024;      // 1M f16
  _Float16* qp   = Wo16 + (size_t)1024 * 1024;      // 32*2048*64 = 4M f16
  _Float16* kp   = qp + (size_t)32 * 2048 * 64;
  _Float16* vp   = kp + (size_t)32 * 2048 * 64;
  _Float16* ctx  = vp + (size_t)32 * 2048 * 64;     // 4096*1024 f16
  float*    Y    = (float*)(ctx + (size_t)4096 * 1024);  // 4096*1024 f32

  float* out_main = (float*)d_out;                           // [2,2048,1024]
  float* w_out    = out_main + (size_t)2 * 2048 * 1024;      // [32,2048,2048]

  mha_cvt_f16<<<4096, 256, 0, stream>>>(Wq,   Wq16, 1024 * 1024);
  mha_cvt_f16<<<4096, 256, 0, stream>>>(Wout, Wo16, 1024 * 1024);

  dim3 gproj(64, 16);  // M/64 x N/64
  mha_proj<<<gproj, 128, 0, stream>>>(q, Wq16, bq, qp);
  mha_proj<<<gproj, 128, 0, stream>>>(k, Wq16, bq, kp);
  mha_proj<<<gproj, 128, 0, stream>>>(v, Wq16, bq, vp);

  dim3 gattn(32, 32);  // S/64 row tiles x (B*H) heads
  mha_attn<<<gattn, 128, 0, stream>>>(qp, kp, vp, mask, w_out, ctx);

  mha_outproj<<<gproj, 128, 0, stream>>>(ctx, Wo16, bout, q, Y);

  mha_layernorm<<<4096, 256, 0, stream>>>(Y, gamma, beta, out_main);
}
